// HAF_7868380086790
// MI455X (gfx1250) — compile-verified, matrix operand order checked by reading, not run on hardware
//
#include <hip/hip_runtime.h>
#include <hip/hip_bf16.h>
#include <math.h>

// ---------------- sizes ----------------
#define NN 64
#define CC 256
#define LL 6
#define TT 64
#define VV 16
#define IC 64          // C/4
#define KK 3
#define LV (LL*VV)     // 96
#define NLV (NN*LV)    // 6144

typedef float v2f __attribute__((ext_vector_type(2)));
typedef float v8f __attribute__((ext_vector_type(8)));
typedef _Float16 v16h __attribute__((ext_vector_type(16)));

#if defined(__has_builtin)
#if __has_builtin(__builtin_amdgcn_wmma_f32_16x16x4_f32)
#define USE_WMMA_F32 1
#endif
#endif

// GROUPS = [[0],[1,10,13],[2,11,14],[3,4,7,12,15],[5,8],[6,9]]
__constant__ int c_goff[LL] = {0, 1, 4, 7, 12, 14};
__constant__ int c_glen[LL] = {1, 3, 3, 5, 2, 2};
__constant__ int c_gidx[16] = {0, 1,10,13, 2,11,14, 3,4,7,12,15, 5,8, 6,9};

// ---------------- K1: x_t = mean over T ----------------
__global__ __launch_bounds__(256) void k_meanT(const float* __restrict__ x,
                                               float* __restrict__ xt) {
  int g = blockIdx.x * 256 + threadIdx.x;           // one thread per (row, v)
  if (g >= NN*CC*LL*VV) return;
  int v   = g & 15;
  int row = g >> 4;                                 // (n*C + c)*L + l
  const float* p = x + (size_t)row * (TT*VV) + v;
  float s = 0.f;
#pragma unroll 8
  for (int t = 0; t < TT; ++t) s += p[t * VV];
  xt[g] = s * (1.0f / (float)TT);
}

// ---------------- K2: h_pre = W_down @ x_t + b_down (WMMA) ----------------
// Per n: D(64 x 96) = W(64 x 256) * X_n(256 x 96). One wave per 16x16 tile.
__global__ __launch_bounds__(32) void k_gemm_down(const float* __restrict__ W,
                                                  const float* __restrict__ Xt,
                                                  const float* __restrict__ bias,
                                                  float* __restrict__ Hpre) {
  int bid = blockIdx.x;                // NN * 4 * 6 tiles
  int nt  = bid % 6;
  int tmp = bid / 6;
  int mt  = tmp % 4;
  int n   = tmp / 4;
  int lane = threadIdx.x;              // wave32
  int half = lane >> 4;                // 0 | 1
  int lid  = lane & 15;
  const float* Xn = Xt + (size_t)n * CC * LV;

  v8f acc = {};
#ifdef USE_WMMA_F32
  // V_WMMA_F32_16X16X4_F32: A 16x4 (2 VGPR), B 4x16 (2 VGPR)
  for (int k0 = 0; k0 < CC; k0 += 4) {
    int ka = k0 + half * 2;
    v2f a, b;
    a.x = W[(size_t)(mt*16 + lid) * CC + ka];
    a.y = W[(size_t)(mt*16 + lid) * CC + ka + 1];
    b.x = Xn[(size_t)ka       * LV + nt*16 + lid];
    b.y = Xn[(size_t)(ka + 1) * LV + nt*16 + lid];
    acc = __builtin_amdgcn_wmma_f32_16x16x4_f32(false, a, false, b,
                                                (short)0, acc, false, false);
  }
#else
  // Fallback: V_WMMA_F32_16X16X32_F16, convert on the fly.
  for (int k0 = 0; k0 < CC; k0 += 32) {
    v16h a, b;
#pragma unroll
    for (int j = 0; j < 16; ++j) {
      int koff = (j < 8) ? (half*8 + j) : (16 + half*8 + (j - 8));
      a[j] = (_Float16)W[(size_t)(mt*16 + lid) * CC + k0 + koff];
      b[j] = (_Float16)Xn[(size_t)(k0 + koff) * LV + nt*16 + lid];
    }
    acc = __builtin_amdgcn_wmma_f32_16x16x32_f16(false, a, false, b,
                                                 (short)0, acc, false, false);
  }
#endif
  // C/D layout: VGPR j -> M = j + half*8, N = lid
#pragma unroll
  for (int j = 0; j < 8; ++j) {
    int o = mt*16 + j + half*8;
    Hpre[((size_t)n * IC + o) * LV + nt*16 + lid] = acc[j] + bias[o];
  }
}

// ---------------- K3/K6: per-channel BN stats -> scale/shift ----------------
// Layout p[(n*64 + o)*J + j]; block per channel o; out[o]=scale, out[64+o]=shift
__global__ __launch_bounds__(256) void k_stats(const float* __restrict__ p,
                                               const float* __restrict__ g,
                                               const float* __restrict__ b,
                                               float* __restrict__ out, int J) {
  __shared__ float s1[256];
  __shared__ float s2[256];
  int o = blockIdx.x, tid = threadIdx.x;
  int total = NN * J;
  float a = 0.f, q = 0.f;
  for (int s = tid; s < total; s += 256) {
    int n = s / J, j = s % J;
    float v = p[((size_t)n * IC + o) * J + j];
    a += v; q += v * v;
  }
  s1[tid] = a; s2[tid] = q;
  __syncthreads();
  for (int st = 128; st > 0; st >>= 1) {
    if (tid < st) { s1[tid] += s1[tid + st]; s2[tid] += s2[tid + st]; }
    __syncthreads();
  }
  if (tid == 0) {
    float inv  = 1.0f / (float)total;
    float mean = s1[0] * inv;
    float var  = s2[0] * inv - mean * mean;
    float sc   = g[o] * rsqrtf(var + 1e-5f);
    out[o]      = sc;
    out[IC + o] = b[o] - mean * sc;
  }
}

// ---------------- K4: BN1 + ReLU + group-sample means -> xs ----------------
__global__ __launch_bounds__(256) void k_bn1_sample(const float* __restrict__ Hpre,
                                                    const float* __restrict__ st,
                                                    float* __restrict__ xs) {
  int e = blockIdx.x * 256 + threadIdx.x;           // (n*64 + o)*6 + l
  if (e >= NN*IC*LL) return;
  int l = e % LL; int t = e / LL; int o = t % IC; int n = t / IC;
  float sc = st[o], sh = st[IC + o];
  const float* hp = Hpre + ((size_t)n * IC + o) * LV + l * VV;
  int off = c_goff[l], len = c_glen[l];
  float acc = 0.f;
  for (int i = 0; i < len; ++i) {
    float v = hp[c_gidx[off + i]] * sc + sh;
    acc += (v > 0.f) ? v : 0.f;
  }
  xs[e] = acc / (float)len;
}

// ---------------- K5: kNN graph + edge GEMM -> e_pre ----------------
__global__ __launch_bounds__(256) void k_edge(const float* __restrict__ xs,
                                              const float* __restrict__ We,
                                              float* __restrict__ Epre) {
  __shared__ float xs_s[IC * LL];   // [o][l]
  __shared__ float d_s[LL * LL];
  __shared__ int   idx_s[LL * KK];
  int n = blockIdx.x, tid = threadIdx.x;
  for (int i = tid; i < IC * LL; i += 256) xs_s[i] = xs[(size_t)n * IC * LL + i];
  __syncthreads();
  if (tid < LL * LL) {              // dist[l][m] = -||x_l - x_m||^2
    int l = tid / LL, m = tid % LL;
    float s = 0.f;
    for (int o = 0; o < IC; ++o) {
      float d = xs_s[o*LL + l] - xs_s[o*LL + m];
      s += d * d;
    }
    d_s[tid] = -s;
  }
  __syncthreads();
  if (tid < LL) {                   // stable top-3 (matches lax.top_k ties)
    int l = tid;
    bool used[LL] = {false,false,false,false,false,false};
    for (int k = 0; k < KK; ++k) {
      int best = 0; float bv = -3.4e38f;
      for (int m = 0; m < LL; ++m)
        if (!used[m] && d_s[l*LL + m] > bv) { bv = d_s[l*LL + m]; best = m; }
      used[best] = true;
      idx_s[l*KK + k] = best;
    }
  }
  __syncthreads();
  for (int e = tid; e < IC * LL * KK; e += 256) {
    int o2 = e / (LL*KK), r = e % (LL*KK), l = r / KK, k = r % KK;
    int m = idx_s[l*KK + k];
    const float* w = We + (size_t)o2 * (2 * IC);
    float acc = 0.f;
    for (int o = 0; o < IC; ++o) {
      float ctr = xs_s[o*LL + l];
      float nb  = xs_s[o*LL + m];
      acc += w[o] * (nb - ctr) + w[IC + o] * ctr;
    }
    Epre[((size_t)n * IC + o2) * (LL*KK) + r] = acc;
  }
}

// ---------------- K7: BN2 + leaky ReLU + max over K -> attmax ----------------
__global__ __launch_bounds__(256) void k_bn2_max(const float* __restrict__ Epre,
                                                 const float* __restrict__ st,
                                                 float* __restrict__ am) {
  int e = blockIdx.x * 256 + threadIdx.x;          // (n*64 + o2)*6 + l
  if (e >= NN*IC*LL) return;
  int l = e % LL; int t = e / LL; int o2 = t % IC; int n = t / IC;
  float sc = st[o2], sh = st[IC + o2];
  const float* ep = Epre + ((size_t)n * IC + o2) * (LL*KK) + l * KK;
  float m = -3.4e38f;
  for (int k = 0; k < KK; ++k) {
    float v = ep[k] * sc + sh;
    v = (v > 0.f) ? v : 0.2f * v;
    m = fmaxf(m, v);
  }
  am[e] = m;
}

// ---------------- K8: att = sigmoid(W_agg @ attmax + b_agg) ----------------
__global__ __launch_bounds__(256) void k_agg(const float* __restrict__ am,
                                             const float* __restrict__ Wa,
                                             const float* __restrict__ ba,
                                             float* __restrict__ att) {
  int e = blockIdx.x * 256 + threadIdx.x;          // (n*256 + co)*6 + l
  if (e >= NN*CC*LL) return;
  int l = e % LL; int t = e / LL; int co = t % CC; int n = t / CC;
  const float* w = Wa + (size_t)co * IC;
  const float* a = am + (size_t)n * IC * LL + l;
  float acc = ba[co];
#pragma unroll 8
  for (int o = 0; o < IC; ++o) acc += w[o] * a[o * LL];
  att[e] = 1.0f / (1.0f + __expf(-acc));
}

// ---------------- K9: out[n,c,t,v] = sum_l x[n,c,l,t,v]*att[n,c,l] ----------
__global__ __launch_bounds__(256) void k_apply(const float* __restrict__ x,
                                               const float* __restrict__ att,
                                               float* __restrict__ out) {
  int bid = blockIdx.x;                 // n*256 + c
  int tid = threadIdx.x;                // one float4 of the 1024-elem (T,V) block
  const float* a = att + (size_t)bid * LL;
  float a0 = a[0], a1 = a[1], a2 = a[2], a3 = a[3], a4 = a[4], a5 = a[5];
  const float4* xp = (const float4*)(x + (size_t)bid * LL * TT * VV);
#pragma unroll
  for (int l = 0; l < LL; ++l)
    __builtin_prefetch(&xp[l * (TT*VV/4) + tid], 0, 1);   // global_prefetch_b8
  float4 v0 = xp[0*256 + tid];
  float4 v1 = xp[1*256 + tid];
  float4 v2 = xp[2*256 + tid];
  float4 v3 = xp[3*256 + tid];
  float4 v4 = xp[4*256 + tid];
  float4 v5 = xp[5*256 + tid];
  float4 r;
  r.x = a0*v0.x + a1*v1.x + a2*v2.x + a3*v3.x + a4*v4.x + a5*v5.x;
  r.y = a0*v0.y + a1*v1.y + a2*v2.y + a3*v3.y + a4*v4.y + a5*v5.y;
  r.z = a0*v0.z + a1*v1.z + a2*v2.z + a3*v3.z + a4*v4.z + a5*v5.z;
  r.w = a0*v0.w + a1*v1.w + a2*v2.w + a3*v3.w + a4*v4.w + a5*v5.w;
  ((float4*)out)[(size_t)bid * 256 + tid] = r;
}

// ---------------- launcher ----------------
extern "C" void kernel_launch(void* const* d_in, const int* in_sizes, int n_in,
                              void* d_out, int out_size, void* d_ws, size_t ws_size,
                              hipStream_t stream) {
  const float* x      = (const float*)d_in[0];
  const float* W_down = (const float*)d_in[1];
  const float* b_down = (const float*)d_in[2];
  const float* bn1_g  = (const float*)d_in[3];
  const float* bn1_b  = (const float*)d_in[4];
  const float* W_edge = (const float*)d_in[5];
  const float* bn2_g  = (const float*)d_in[6];
  const float* bn2_b  = (const float*)d_in[7];
  const float* W_agg  = (const float*)d_in[8];
  const float* b_agg  = (const float*)d_in[9];
  float* out = (float*)d_out;

  float* ws   = (float*)d_ws;
  float* xt   = ws;                                  // N*C*L*V   = 1572864
  float* hpre = xt   + (size_t)NN*CC*LL*VV;          // N*64*96   =  393216
  float* st1  = hpre + (size_t)NN*IC*LV;             // 128
  float* xs   = st1  + 128;                          // N*64*6    =   24576
  float* epre = xs   + (size_t)NN*IC*LL;             // N*64*18   =   73728
  float* st2  = epre + (size_t)NN*IC*LL*KK;          // 128
  float* am   = st2  + 128;                          // N*64*6    =   24576
  float* att  = am   + (size_t)NN*IC*LL;             // N*256*6   =   98304

  k_meanT     <<<(NN*CC*LL*VV + 255)/256, 256, 0, stream>>>(x, xt);
  k_gemm_down <<<NN*4*6, 32, 0, stream>>>(W_down, xt, b_down, hpre);
  k_stats     <<<IC, 256, 0, stream>>>(hpre, bn1_g, bn1_b, st1, LV);
  k_bn1_sample<<<(NN*IC*LL + 255)/256, 256, 0, stream>>>(hpre, st1, xs);
  k_edge      <<<NN, 256, 0, stream>>>(xs, W_edge, epre);
  k_stats     <<<IC, 256, 0, stream>>>(epre, bn2_g, bn2_b, st2, LL*KK);
  k_bn2_max   <<<(NN*IC*LL + 255)/256, 256, 0, stream>>>(epre, st2, am);
  k_agg       <<<(NN*CC*LL + 255)/256, 256, 0, stream>>>(am, W_agg, b_agg, att);
  k_apply     <<<NN*CC, 256, 0, stream>>>(x, att, out);
}
